// smamba_52699248722201
// MI455X (gfx1250) — matrix operand order repaired and checked by reading, not compiled, not run on hardware
//
#include <hip/hip_runtime.h>
#include <hip/hip_bf16.h>
#include <math.h>

typedef __attribute__((ext_vector_type(16))) _Float16 v16h;
typedef __attribute__((ext_vector_type(8)))  _Float16 v8h;
typedef __attribute__((ext_vector_type(8)))  float    v8f;

#define NB      2
#define SEQ     2048
#define DMODEL  512
#define DINNER  512
#define DSTATE  32
#define DTRANK  32
#define MROWS   (NB * SEQ)          /* 4096 rows of (b,l) */
#define DXZ     (2 * DINNER)        /* 1024 */
#define DBLW    (DTRANK + 2*DSTATE) /* 96   */

// ---------------------------------------------------------------------------
// f32 -> f16 conversion (weights once per direction; x once)
// ---------------------------------------------------------------------------
__global__ void smamba_cvt(const float* __restrict__ src,
                           _Float16* __restrict__ dst, int n) {
  const int i = blockIdx.x * 256 + threadIdx.x;
  if (i < n) dst[i] = (_Float16)src[i];
}

// ---------------------------------------------------------------------------
// WMMA 16x16x32 f16 tile engine (ISA 7.12.2 layouts, wave32).
// arow points at A + row(lane)*lda + ((lane>>4)<<3)  : per-lane K offset 0/8.
// A frag per lane: halves {k0+0..7, k0+16..23} (lanes 0-15) / {+8..15,+24..31}.
// B frag per lane: n = lane&15, same K halves as A (koff 0/8, +16).
// NBT = number of 16-wide N tiles per wave (A fragment reuse factor).
// Software-pipelined one K-step deep: next-step fragments are loaded before
// the current WMMA pack issues, so waits become partial instead of loadcnt==0.
// ---------------------------------------------------------------------------
__device__ __forceinline__ v16h ld_frag(const _Float16* __restrict__ p, int off2) {
  v8h x0 = *(const v8h*)p;
  v8h x1 = *(const v8h*)(p + off2);
  v16h r;
#pragma unroll
  for (int i = 0; i < 8; ++i) { r[i] = x0[i]; r[8 + i] = x1[i]; }
  return r;
}

template <int NBT>
__device__ __forceinline__ void gemm_tiles(const _Float16* __restrict__ arow,
                                           const _Float16* __restrict__ W, int ldw,
                                           int n0, int K, int lane, v8f (&c)[NBT]) {
  const int n     = lane & 15;
  const int koffb = (lane >> 4) << 3;          // 0 or 8
  const _Float16* wbase = W + (size_t)(n0 + n) * ldw + koffb;

  // prologue: fragments for k0 = 0
  v16h a = ld_frag(arow, 16);
  v16h b[NBT];
#pragma unroll
  for (int u = 0; u < NBT; ++u) b[u] = ld_frag(wbase + (size_t)(16 * u) * ldw, 16);

  for (int k0 = 0; k0 < K; k0 += 32) {
    const int kn = (k0 + 32 < K) ? (k0 + 32) : k0;   // branch-free tail clamp
    __builtin_prefetch(arow + k0 + 256, 0, 0);       // global_prefetch_b8

    // issue next-step loads before this step's WMMAs (latency overlap)
    v16h an = ld_frag(arow + kn, 16);
    v16h bn[NBT];
#pragma unroll
    for (int u = 0; u < NBT; ++u)
      bn[u] = ld_frag(wbase + (size_t)(16 * u) * ldw + kn, 16);

#pragma unroll
    for (int u = 0; u < NBT; ++u)
      c[u] = __builtin_amdgcn_wmma_f32_16x16x32_f16(false, a, false, b[u],
                                                    (short)0, c[u], false, false);
    a = an;
#pragma unroll
    for (int u = 0; u < NBT; ++u) b[u] = bn[u];
  }
}

// ---------------------------------------------------------------------------
// 1) in-projection: xz[r,0:1024] = x[rs,:] @ in_w^T  (rs = flipped row)
//    16x64 tile per wave; tiles = 256 m * 16 nblk = 4096 waves
// ---------------------------------------------------------------------------
__global__ void smamba_inproj(const _Float16* __restrict__ xh,
                              const _Float16* __restrict__ in_wh,
                              float* __restrict__ xz, int flip) {
  const int lane = threadIdx.x & 31;
  const int tile = blockIdx.x * 8 + (threadIdx.x >> 5);
  const int mt = tile >> 4, nblk = tile & 15;
  const int m0 = mt * 16, n0 = nblk * 64;

  const int r  = m0 + (lane & 15);
  const int b  = r / SEQ, l = r % SEQ;
  const int rs = flip ? (b * SEQ + (SEQ - 1 - l)) : r;
  const _Float16* arow = xh + (size_t)rs * DMODEL + ((lane >> 4) << 3);

  v8f c[4] = {};
  gemm_tiles<4>(arow, in_wh, DMODEL, n0, DMODEL, lane, c);

  const int n = lane & 15, mb = (lane >> 4) << 3;
#pragma unroll
  for (int u = 0; u < 4; ++u)
#pragma unroll
    for (int j = 0; j < 8; ++j)
      xz[(size_t)(m0 + mb + j) * DXZ + n0 + 16 * u + n] = c[u][j];
}

// ---------------------------------------------------------------------------
// 2) depthwise conv (K=2) + SiLU; writes f32 xc and its f16 mirror xch
// ---------------------------------------------------------------------------
__global__ void smamba_conv(const float* __restrict__ xz,
                            const float* __restrict__ conv_w,
                            const float* __restrict__ conv_b,
                            float* __restrict__ xc,
                            _Float16* __restrict__ xch) {
  const int idx = blockIdx.x * blockDim.x + threadIdx.x;
  if (idx >= MROWS * DINNER) return;
  const int r = idx / DINNER, d = idx - r * DINNER;
  const int l = r % SEQ;
  const float prev = (l == 0) ? 0.f : xz[(size_t)(r - 1) * DXZ + d];
  const float cur  = xz[(size_t)r * DXZ + d];
  float v = conv_b[d] + conv_w[2 * d] * prev + conv_w[2 * d + 1] * cur;
  v = v / (1.f + __expf(-v));                       // SiLU
  xc [(size_t)r * DINNER + d] = v;
  xch[(size_t)r * DINNER + d] = (_Float16)v;
}

// ---------------------------------------------------------------------------
// 3) x-projection: dbl[r,0:96] = xc @ xproj_w^T ; also f16 mirror dblh
//    16x48 tile per wave (NBT=3); tiles = 256 m * 2 nblk = 512 waves
// ---------------------------------------------------------------------------
__global__ void smamba_xproj(const _Float16* __restrict__ xch,
                             const _Float16* __restrict__ xproj_wh,
                             float* __restrict__ dbl,
                             _Float16* __restrict__ dblh) {
  const int lane = threadIdx.x & 31;
  const int tile = blockIdx.x * 8 + (threadIdx.x >> 5);
  const int mt = tile >> 1, nblk = tile & 1;
  const int m0 = mt * 16, n0 = nblk * 48;

  const _Float16* arow = xch + (size_t)(m0 + (lane & 15)) * DINNER + ((lane >> 4) << 3);
  v8f c[3] = {};
  gemm_tiles<3>(arow, xproj_wh, DINNER, n0, DINNER, lane, c);

  const int n = lane & 15, mb = (lane >> 4) << 3;
#pragma unroll
  for (int u = 0; u < 3; ++u)
#pragma unroll
    for (int j = 0; j < 8; ++j) {
      const size_t o = (size_t)(m0 + mb + j) * DBLW + n0 + 16 * u + n;
      dbl [o] = c[u][j];
      dblh[o] = (_Float16)c[u][j];
    }
}

// ---------------------------------------------------------------------------
// 4) dt[r,0:512] = softplus(dbl[r,0:32] @ dt_w^T + dt_b)   (K=32: 1 WMMA/tile)
//    16x64 per wave; tiles = 256 m * 8 nblk = 2048 waves
// ---------------------------------------------------------------------------
__global__ void smamba_dt(const _Float16* __restrict__ dblh,
                          const _Float16* __restrict__ dt_wh,
                          const float* __restrict__ dt_b,
                          float* __restrict__ dt) {
  const int lane = threadIdx.x & 31;
  const int tile = blockIdx.x * 8 + (threadIdx.x >> 5);
  const int mt = tile >> 3, nblk = tile & 7;
  const int m0 = mt * 16, n0 = nblk * 64;

  const _Float16* arow = dblh + (size_t)(m0 + (lane & 15)) * DBLW + ((lane >> 4) << 3);
  v8f c[4] = {};
  gemm_tiles<4>(arow, dt_wh, DTRANK, n0, DTRANK, lane, c);

  const int n = lane & 15, mb = (lane >> 4) << 3;
#pragma unroll
  for (int u = 0; u < 4; ++u) {
    const float bias = dt_b[n0 + 16 * u + n];
#pragma unroll
    for (int j = 0; j < 8; ++j) {
      float v = c[u][j] + bias;
      v = (v > 20.f) ? v : log1pf(__expf(v));        // softplus
      dt[(size_t)(m0 + mb + j) * DINNER + n0 + 16 * u + n] = v;
    }
  }
}

// ---------------------------------------------------------------------------
// 5) selective scan: one wave per (b,d) channel, lane n = state index.
//    h[n] <- exp(dt*A[d,n])*h[n] + (dt*xc)*B[t,n] ;  y = sum_n h[n]*C[t,n]
//    gated output written straight to f16 for the out-projection.
// ---------------------------------------------------------------------------
__global__ void smamba_scan(const float* __restrict__ xz,
                            const float* __restrict__ xc,
                            const float* __restrict__ dbl,
                            const float* __restrict__ dt,
                            const float* __restrict__ A_log,
                            const float* __restrict__ Dp,
                            _Float16* __restrict__ yh) {
  const int gw   = (blockIdx.x * blockDim.x + threadIdx.x) >> 5;
  const int lane = threadIdx.x & 31;
  if (gw >= NB * DINNER) return;
  const int b = gw / DINNER, d = gw - b * DINNER;

  const float Adn = -__expf(A_log[d * DSTATE + lane]);
  const float Dpd = Dp[d];
  float h = 0.f;
  const size_t base = (size_t)b * SEQ;

  for (int t = 0; t < SEQ; ++t) {
    const size_t r = base + t;
    const float dtv = dt[r * DINNER + d];            // wave-uniform broadcast
    const float xcv = xc[r * DINNER + d];
    const float Bv  = dbl[r * DBLW + DTRANK + lane];           // per-lane n
    const float Cv  = dbl[r * DBLW + DTRANK + DSTATE + lane];
    __builtin_prefetch(&dbl[(r + 2) * DBLW + DTRANK + lane], 0, 0);

    h = __expf(dtv * Adn) * h + (dtv * xcv) * Bv;
    float p = h * Cv;
#pragma unroll
    for (int o = 16; o > 0; o >>= 1) p += __shfl_xor(p, o, 32);

    if (lane == 0) {
      const float zv = xz[r * DXZ + DINNER + d];
      const float yv = (p + xcv * Dpd) * (zv / (1.f + __expf(-zv)));
      yh[r * DINNER + d] = (_Float16)yv;
    }
  }
}

// ---------------------------------------------------------------------------
// 6) out-projection: out[ro,:] (+)= y[r,:] @ out_w^T  (ro = flipped row)
//    16x64 per wave; tiles = 256 m * 8 nblk = 2048 waves
// ---------------------------------------------------------------------------
__global__ void smamba_outproj(const _Float16* __restrict__ yh,
                               const _Float16* __restrict__ out_wh,
                               float* __restrict__ out, int flip, int accum) {
  const int lane = threadIdx.x & 31;
  const int tile = blockIdx.x * 8 + (threadIdx.x >> 5);
  const int mt = tile >> 3, nblk = tile & 7;
  const int m0 = mt * 16, n0 = nblk * 64;

  const _Float16* arow = yh + (size_t)(m0 + (lane & 15)) * DINNER + ((lane >> 4) << 3);
  v8f c[4] = {};
  gemm_tiles<4>(arow, out_wh, DINNER, n0, DINNER, lane, c);

  const int n = lane & 15, mb = (lane >> 4) << 3;
#pragma unroll
  for (int u = 0; u < 4; ++u)
#pragma unroll
    for (int j = 0; j < 8; ++j) {
      const int r = m0 + mb + j;
      const int b = r / SEQ, l = r % SEQ;
      const int ro = flip ? (b * SEQ + (SEQ - 1 - l)) : r;
      const size_t o = (size_t)ro * DMODEL + n0 + 16 * u + n;
      if (accum) out[o] += c[u][j]; else out[o] = c[u][j];
    }
}

// ---------------------------------------------------------------------------
// launch
// ---------------------------------------------------------------------------
static inline size_t align_up(size_t v, size_t a) { return (v + a - 1) & ~(a - 1); }

extern "C" void kernel_launch(void* const* d_in, const int* in_sizes, int n_in,
                              void* d_out, int out_size, void* d_ws, size_t ws_size,
                              hipStream_t stream) {
  const float* x = (const float*)d_in[0];
  float* out = (float*)d_out;

  // ---- workspace carve-up (256B-aligned slabs) ----
  char* w = (char*)d_ws;
  auto take = [&](size_t bytes) { char* p = w; w += align_up(bytes, 256); return p; };
  float*    xz    = (float*)   take((size_t)MROWS * DXZ    * 4);  // 16 MB
  float*    xc    = (float*)   take((size_t)MROWS * DINNER * 4);  //  8 MB
  float*    dbl   = (float*)   take((size_t)MROWS * DBLW   * 4);  //  1.5 MB
  float*    dt    = (float*)   take((size_t)MROWS * DINNER * 4);  //  8 MB
  _Float16* xh    = (_Float16*)take((size_t)MROWS * DMODEL * 2);  //  4 MB
  _Float16* xch   = (_Float16*)take((size_t)MROWS * DINNER * 2);  //  4 MB
  _Float16* dblh  = (_Float16*)take((size_t)MROWS * DBLW   * 2);  //  0.75 MB
  _Float16* yh    = (_Float16*)take((size_t)MROWS * DINNER * 2);  //  4 MB
  _Float16* in_wh = (_Float16*)take((size_t)DXZ * DMODEL   * 2);  //  1 MB
  _Float16* xp_wh = (_Float16*)take((size_t)DBLW * DINNER  * 2);  //  96 KB
  _Float16* dt_wh = (_Float16*)take((size_t)DINNER * DTRANK* 2);  //  32 KB
  _Float16* o_wh  = (_Float16*)take((size_t)DMODEL * DINNER* 2);  //  0.5 MB

  // x -> f16 once (flip handled by row remap at the A-fragment load)
  smamba_cvt<<<(MROWS * DMODEL + 255) / 256, 256, 0, stream>>>(x, xh, MROWS * DMODEL);

  for (int dir = 0; dir < 2; ++dir) {
    const float* const* p = (const float* const*)(d_in + 1 + dir * 9);
    const float* in_w    = p[0];
    const float* conv_w  = p[1];
    const float* conv_b  = p[2];
    const float* xproj_w = p[3];
    const float* dt_w    = p[4];
    const float* dt_b    = p[5];
    const float* A_log   = p[6];
    const float* Dp      = p[7];
    const float* out_w   = p[8];

    smamba_cvt<<<(DXZ * DMODEL + 255) / 256,     256, 0, stream>>>(in_w,    in_wh, DXZ * DMODEL);
    smamba_cvt<<<(DBLW * DINNER + 255) / 256,    256, 0, stream>>>(xproj_w, xp_wh, DBLW * DINNER);
    smamba_cvt<<<(DINNER * DTRANK + 255) / 256,  256, 0, stream>>>(dt_w,    dt_wh, DINNER * DTRANK);
    smamba_cvt<<<(DMODEL * DINNER + 255) / 256,  256, 0, stream>>>(out_w,   o_wh,  DMODEL * DINNER);

    smamba_inproj <<<512, 256, 0, stream>>>(xh, in_wh, xz, dir);
    smamba_conv   <<<(MROWS * DINNER) / 256, 256, 0, stream>>>(xz, conv_w, conv_b, xc, xch);
    smamba_xproj  <<<64,  256, 0, stream>>>(xch, xp_wh, dbl, dblh);
    smamba_dt     <<<256, 256, 0, stream>>>(dblh, dt_wh, dt_b, dt);
    smamba_scan   <<<128, 256, 0, stream>>>(xz, xc, dbl, dt, A_log, Dp, yh);
    smamba_outproj<<<256, 256, 0, stream>>>(yh, o_wh, out, dir, dir);
  }
}
